// MultiHeadAttention_27419071218139
// MI455X (gfx1250) — matrix operand order repaired and checked by reading, compiled, not run on hardware
//
#include <hip/hip_runtime.h>
#include <math.h>

// ---------------------------------------------------------------------------
// MHA forward on gfx1250 (wave32, WMMA). All GEMMs via v_wmma_f32_16x16x32_bf16.
// Compute-bound op (~69 GFLOP vs ~100MB HBM) -> maximize WMMA issue density.
// ---------------------------------------------------------------------------

typedef __attribute__((ext_vector_type(16))) __bf16 v16bf;
typedef __attribute__((ext_vector_type(8)))  float  v8f;

#define B_SZ 2
#define SEQ  2048
#define DIM  1024
#define NH   16
#define HD   64
#define N3   3072
#define MTOT (B_SZ * SEQ)   // 4096 rows for the projection GEMMs

// fp32 -> bf16, round-to-nearest-even
__device__ __forceinline__ unsigned short f2bf(float x) {
  unsigned int u = __float_as_uint(x);
  u += 0x7FFFu + ((u >> 16) & 1u);
  return (unsigned short)(u >> 16);
}

__device__ __forceinline__ v8f wmma_bf16(v16bf a, v16bf b, v8f c) {
  // 8 args: (neg_a, A, neg_b, B, c_mod, C, reuse_a, reuse_b)
  return __builtin_amdgcn_wmma_f32_16x16x32_bf16(false, a, false, b, (short)0, c,
                                                 false, false);
}

// A-matrix 16x32 bf16 fragment (ISA 7.12.2): lanes 0-15 -> rows, lower half
// holds K in {0..7, 16..23}, upper half +8. LDS tile layout [row][K], K dense.
__device__ __forceinline__ v16bf frag_ld_a(const unsigned short* base, int row0,
                                           int stride, int k0) {
  const int lane = threadIdx.x & 31;
  const unsigned short* rp =
      base + (row0 + (lane & 15)) * stride + k0 + ((lane >> 4) << 3);
  union { v16bf v; unsigned int u[8]; } f;
#pragma unroll
  for (int i = 0; i < 8; ++i) {
    const int kk = ((i & 3) << 1) + ((i & 4) << 2);  // 0,2,4,6,16,18,20,22
    f.u[i] = *(const unsigned int*)(rp + kk);
  }
  return f.v;
}

// B-matrix 32x16 bf16 fragment: lanes 0-15 -> columns, lower half holds
// K in {0..7, 16..23}, upper half +8 (same as A). LDS layout [n][K].
__device__ __forceinline__ v16bf frag_ld_b(const unsigned short* base, int row0,
                                           int stride, int k0) {
  const int lane = threadIdx.x & 31;
  const unsigned short* rp =
      base + (row0 + (lane & 15)) * stride + k0 + ((lane >> 4) << 3);
  union { v16bf v; unsigned int u[8]; } f;
#pragma unroll
  for (int i = 0; i < 8; ++i) f.u[i] = *(const unsigned int*)(rp + ((i & 3) << 1) + ((i & 4) << 2));
  return f.v;
}

// ---------------------------------------------------------------------------
// Stage 1: fp32 -> bf16 conversion (sizes are all multiples of 4)
// ---------------------------------------------------------------------------
__global__ void cvt_bf16(const float* __restrict__ in,
                         unsigned short* __restrict__ out, int n) {
  int i = (blockIdx.x * blockDim.x + threadIdx.x) * 4;
  if (i + 3 < n) {
    float4 f = *(const float4*)(in + i);
    uint2 o;
    o.x = (unsigned int)f2bf(f.x) | ((unsigned int)f2bf(f.y) << 16);
    o.y = (unsigned int)f2bf(f.z) | ((unsigned int)f2bf(f.w) << 16);
    *(uint2*)(out + i) = o;
  }
}

// ---------------------------------------------------------------------------
// Generic GEMM body: block tile 128x64, 8 waves in 4x2 grid, each wave a
// 32x32 output tile -> 4 WMMAs per K-step with 2x fragment reuse.
// A: [M,DIM] bf16 row-major; B: [DIM,N] bf16 row-major, staged transposed.
// ---------------------------------------------------------------------------
#define GEMM_BODY(Aptr, Bptr, ldb)                                             \
  __shared__ unsigned short As[128 * 36]; /* [row][K] */                       \
  __shared__ unsigned short Bs[64 * 36];  /* [n][K]   */                       \
  const int tid = threadIdx.x;                                                 \
  const int w = tid >> 5;                                                      \
  const int mw = w & 3, nw = w >> 2;                                           \
  v8f acc00 = {}, acc01 = {}, acc10 = {}, acc11 = {};                          \
  for (int kt = 0; kt < DIM; kt += 32) {                                       \
    _Pragma("unroll") for (int j = 0; j < 8; ++j) { /* A: 2048 dwords */       \
      int u = tid + j * 256;                                                   \
      int row = u >> 4, kp = (u & 15) << 1;                                    \
      *(unsigned int*)(As + row * 36 + kp) =                                   \
          *(const unsigned int*)(Aptr + (size_t)(m0 + row) * DIM + kt + kp);   \
    }                                                                          \
    _Pragma("unroll") for (int j = 0; j < 8; ++j) { /* B: 2048 bf16, transp */ \
      int u = tid + j * 256;                                                   \
      int kk = u >> 6, nn = u & 63;                                            \
      Bs[nn * 36 + kk] = Bptr[(size_t)(kt + kk) * (ldb) + n0 + nn];            \
    }                                                                          \
    if (kt + 32 < DIM) { /* gfx1250 global_prefetch_b8 of next K-tile */       \
      __builtin_prefetch(Aptr + (size_t)(m0 + (tid >> 1)) * DIM + kt + 32, 0, 0); \
      __builtin_prefetch(Bptr + (size_t)(kt + 32 + (tid >> 3)) * (ldb) + n0, 0, 0); \
    }                                                                          \
    __syncthreads();                                                           \
    v16bf a0 = frag_ld_a(As, (mw * 2 + 0) * 16, 36, 0);                        \
    v16bf a1 = frag_ld_a(As, (mw * 2 + 1) * 16, 36, 0);                        \
    v16bf b0 = frag_ld_b(Bs, (nw * 2 + 0) * 16, 36, 0);                        \
    v16bf b1 = frag_ld_b(Bs, (nw * 2 + 1) * 16, 36, 0);                        \
    acc00 = wmma_bf16(a0, b0, acc00);                                          \
    acc01 = wmma_bf16(a0, b1, acc01);                                          \
    acc10 = wmma_bf16(a1, b0, acc10);                                          \
    acc11 = wmma_bf16(a1, b1, acc11);                                          \
    __syncthreads();                                                           \
  }                                                                            \
  const int lane = tid & 31, half = lane >> 4, nrel = lane & 15;

// ---------------------------------------------------------------------------
// Stage 2: QKV projection [4096,1024] x [1024,3072] + bias, scattered into
// Q/K/V bf16 buffers laid out [B][H][S][HD].
// ---------------------------------------------------------------------------
__global__ __launch_bounds__(256) void qkv_gemm(
    const unsigned short* __restrict__ xb, const unsigned short* __restrict__ wb,
    const float* __restrict__ bias, unsigned short* __restrict__ qo,
    unsigned short* __restrict__ ko, unsigned short* __restrict__ vo) {
  const int m0 = (blockIdx.x / 48) * 128;
  const int n0 = (blockIdx.x % 48) * 64;
  GEMM_BODY(xb, wb, N3)
#pragma unroll
  for (int i = 0; i < 2; ++i) {
#pragma unroll
    for (int j = 0; j < 2; ++j) {
      v8f acc = (i == 0) ? (j == 0 ? acc00 : acc01) : (j == 0 ? acc10 : acc11);
      int col = n0 + (nw * 2 + j) * 16 + nrel;
      int h = col / 192, rem = col % 192;  // 3*HD = 192
      int sel = rem / HD, d = rem % HD;
      unsigned short* dst = (sel == 0) ? qo : ((sel == 1) ? ko : vo);
      float bv = bias[col];
#pragma unroll
      for (int r = 0; r < 8; ++r) {
        int row = m0 + (mw * 2 + i) * 16 + r + half * 8;
        int bb = row >> 11, s = row & (SEQ - 1);
        dst[(((bb * NH + h) * SEQ) + s) * HD + d] = f2bf(acc[r] + bv);
      }
    }
  }
}

// ---------------------------------------------------------------------------
// Stage 3: flash attention. One block = 128 q-rows of one (b,h).
// Wave w owns q-rows [w*16, w*16+16) so online-softmax state stays in-wave.
// ---------------------------------------------------------------------------
__global__ __launch_bounds__(256) void attn(
    const unsigned short* __restrict__ q, const unsigned short* __restrict__ k,
    const unsigned short* __restrict__ v, unsigned short* __restrict__ o) {
  __shared__ unsigned short Qs[128 * 68];     // [qrow][hd]
  __shared__ unsigned short Ks[64 * 68];      // [key][hd]
  __shared__ unsigned short Vs[64 * 68];      // [hd][key] (transposed)
  __shared__ unsigned short Ps[8 * 16 * 68];  // per-wave P tile [qrow][key]
  const int tid = threadIdx.x;
  const int w = tid >> 5;
  const int lane = tid & 31;
  const int half = lane >> 4;
  const int qt = blockIdx.x & 15;
  const int h = (blockIdx.x >> 4) & 15;
  const int b = blockIdx.x >> 8;
  const unsigned short* qb = q + ((size_t)(b * NH + h) * SEQ + qt * 128) * HD;
  const unsigned short* kb = k + (size_t)(b * NH + h) * SEQ * HD;
  const unsigned short* vb = v + (size_t)(b * NH + h) * SEQ * HD;

#pragma unroll
  for (int j = 0; j < 16; ++j) {  // Q tile: 4096 dwords
    int u = tid + j * 256;
    int row = u >> 5, kp = (u & 31) << 1;
    *(unsigned int*)(Qs + row * 68 + kp) =
        *(const unsigned int*)(qb + row * HD + kp);
  }

  v8f O0 = {}, O1 = {}, O2 = {}, O3 = {};
  float mrow[8], lrow[8];
#pragma unroll
  for (int r = 0; r < 8; ++r) { mrow[r] = -1e30f; lrow[r] = 0.f; }

  for (int kt = 0; kt < SEQ; kt += 64) {
    __syncthreads();  // protect Ks/Vs (and initial Qs)
#pragma unroll
    for (int j = 0; j < 8; ++j) {  // K tile: 2048 dwords
      int u = tid + j * 256;
      int row = u >> 5, kp = (u & 31) << 1;
      *(unsigned int*)(Ks + row * 68 + kp) =
          *(const unsigned int*)(kb + (kt + row) * HD + kp);
    }
#pragma unroll
    for (int j = 0; j < 16; ++j) {  // V tile, transposed to [hd][key]
      int u = tid + j * 256;
      int kk = u >> 6, dd = u & 63;
      Vs[dd * 68 + kk] = vb[(kt + kk) * HD + dd];
    }
    if (kt + 64 < SEQ) {  // prefetch next K/V tiles into GL2
      __builtin_prefetch(kb + (size_t)(kt + 64 + (tid >> 2)) * HD, 0, 0);
      __builtin_prefetch(vb + (size_t)(kt + 64 + (tid >> 2)) * HD, 0, 0);
    }
    __syncthreads();

    // S = Q K^T over Hd=64 (two K=32 WMMA steps), 16x64 per wave
    v16bf a0 = frag_ld_a(Qs, w * 16, 68, 0);
    v16bf a1 = frag_ld_a(Qs, w * 16, 68, 32);
    v8f sc[4];
#pragma unroll
    for (int n = 0; n < 4; ++n) {
      v8f t = {};
      t = wmma_bf16(a0, frag_ld_b(Ks, n * 16, 68, 0), t);
      t = wmma_bf16(a1, frag_ld_b(Ks, n * 16, 68, 32), t);
      sc[n] = t;
    }

    const float scale = 0.125f;  // 1/sqrt(64)
    float mnew[8], alpha[8], rsum[8];
#pragma unroll
    for (int r = 0; r < 8; ++r) {
      float m = -1e30f;
#pragma unroll
      for (int n = 0; n < 4; ++n) m = fmaxf(m, sc[n][r]);
#pragma unroll
      for (int off = 8; off >= 1; off >>= 1)  // reduce across 16-lane half
        m = fmaxf(m, __shfl_xor(m, off, 32));
      m *= scale;
      float mn = fmaxf(mrow[r], m);
      mnew[r] = mn;
      alpha[r] = __expf(mrow[r] - mn);
      rsum[r] = 0.f;
    }

    // P = exp(S*scale - mnew) into this wave's private LDS region
    unsigned short* Pw = Ps + w * 16 * 68;
#pragma unroll
    for (int n = 0; n < 4; ++n) {
#pragma unroll
      for (int r = 0; r < 8; ++r) {
        float p = __expf(sc[n][r] * scale - mnew[r]);
        rsum[r] += p;
        Pw[(r + half * 8) * 68 + n * 16 + (lane & 15)] = f2bf(p);
      }
    }
#pragma unroll
    for (int r = 0; r < 8; ++r) {
      float s = rsum[r];
#pragma unroll
      for (int off = 8; off >= 1; off >>= 1) s += __shfl_xor(s, off, 32);
      lrow[r] = lrow[r] * alpha[r] + s;
      mrow[r] = mnew[r];
    }
#pragma unroll
    for (int r = 0; r < 8; ++r) {  // rescale running O
      O0[r] *= alpha[r]; O1[r] *= alpha[r]; O2[r] *= alpha[r]; O3[r] *= alpha[r];
    }

    // O += P @ V (same-wave LDS write->read; DS ops in-order per wave)
    v16bf p0 = frag_ld_a(Pw, 0, 68, 0);
    v16bf p1 = frag_ld_a(Pw, 0, 68, 32);
    O0 = wmma_bf16(p0, frag_ld_b(Vs, 0, 68, 0), O0);
    O0 = wmma_bf16(p1, frag_ld_b(Vs, 0, 68, 32), O0);
    O1 = wmma_bf16(p0, frag_ld_b(Vs, 16, 68, 0), O1);
    O1 = wmma_bf16(p1, frag_ld_b(Vs, 16, 68, 32), O1);
    O2 = wmma_bf16(p0, frag_ld_b(Vs, 32, 68, 0), O2);
    O2 = wmma_bf16(p1, frag_ld_b(Vs, 32, 68, 32), O2);
    O3 = wmma_bf16(p0, frag_ld_b(Vs, 48, 68, 0), O3);
    O3 = wmma_bf16(p1, frag_ld_b(Vs, 48, 68, 32), O3);
  }

  const int nrel = lane & 15;
#pragma unroll
  for (int r = 0; r < 8; ++r) {
    float inv = 1.f / lrow[r];
    int srow = qt * 128 + w * 16 + r + half * 8;
    unsigned short* orow = o + ((size_t)b * SEQ + srow) * DIM + h * HD;
    orow[0 + nrel]  = f2bf(O0[r] * inv);
    orow[16 + nrel] = f2bf(O1[r] * inv);
    orow[32 + nrel] = f2bf(O2[r] * inv);
    orow[48 + nrel] = f2bf(O3[r] * inv);
  }
}

// ---------------------------------------------------------------------------
// Stage 4: out projection [4096,1024] x [1024,1024] + bias -> fp32 d_out
// ---------------------------------------------------------------------------
__global__ __launch_bounds__(256) void out_gemm(
    const unsigned short* __restrict__ ab, const unsigned short* __restrict__ wb,
    const float* __restrict__ bias, float* __restrict__ out) {
  const int m0 = (blockIdx.x >> 4) * 128;
  const int n0 = (blockIdx.x & 15) * 64;
  GEMM_BODY(ab, wb, DIM)
#pragma unroll
  for (int i = 0; i < 2; ++i) {
#pragma unroll
    for (int j = 0; j < 2; ++j) {
      v8f acc = (i == 0) ? (j == 0 ? acc00 : acc01) : (j == 0 ? acc10 : acc11);
      int col = n0 + (nw * 2 + j) * 16 + nrel;
      float bv = bias[col];
#pragma unroll
      for (int r = 0; r < 8; ++r) {
        int row = m0 + (mw * 2 + i) * 16 + r + half * 8;
        out[(size_t)row * DIM + col] = acc[r] + bv;
      }
    }
  }
}

// ---------------------------------------------------------------------------
extern "C" void kernel_launch(void* const* d_in, const int* in_sizes, int n_in,
                              void* d_out, int out_size, void* d_ws,
                              size_t ws_size, hipStream_t stream) {
  (void)in_sizes; (void)n_in; (void)out_size; (void)ws_size;
  const float* x     = (const float*)d_in[0];
  const float* w_qkv = (const float*)d_in[1];
  const float* b_qkv = (const float*)d_in[2];
  const float* w_out = (const float*)d_in[3];
  const float* b_out = (const float*)d_in[4];
  float* out = (float*)d_out;

  // Workspace layout (bf16 = u16), total 24M elems = 48 MB
  unsigned short* xb    = (unsigned short*)d_ws;
  unsigned short* wqkvb = xb    + (size_t)MTOT * DIM;  // 4M
  unsigned short* woutb = wqkvb + (size_t)DIM * N3;    // 3M
  unsigned short* qbuf  = woutb + (size_t)DIM * DIM;   // 1M
  unsigned short* kbuf  = qbuf  + (size_t)MTOT * DIM;  // 4M
  unsigned short* vbuf  = kbuf  + (size_t)MTOT * DIM;  // 4M
  unsigned short* abuf  = vbuf  + (size_t)MTOT * DIM;  // 4M (+4M)

  cvt_bf16<<<(MTOT * DIM / 4 + 255) / 256, 256, 0, stream>>>(x, xb, MTOT * DIM);
  cvt_bf16<<<(DIM * N3 / 4 + 255) / 256, 256, 0, stream>>>(w_qkv, wqkvb, DIM * N3);
  cvt_bf16<<<(DIM * DIM / 4 + 255) / 256, 256, 0, stream>>>(w_out, woutb, DIM * DIM);

  qkv_gemm<<<(MTOT / 128) * (N3 / 64), 256, 0, stream>>>(xb, wqkvb, b_qkv, qbuf,
                                                         kbuf, vbuf);
  attn<<<B_SZ * NH * (SEQ / 128), 256, 0, stream>>>(qbuf, kbuf, vbuf, abuf);
  out_gemm<<<(MTOT / 128) * (DIM / 64), 256, 0, stream>>>(abuf, woutb, b_out, out);
}
